// myRGCN_87703232184764
// MI455X (gfx1250) — matrix operand order repaired and checked by reading, not compiled, not run on hardware
//
#include <hip/hip_runtime.h>
#include <hip/hip_bf16.h>

typedef __attribute__((ext_vector_type(16))) _Float16 v16h;
typedef __attribute__((ext_vector_type(8)))  _Float16 v8h;
typedef __attribute__((ext_vector_type(8)))  float    v8f;

#define NNODES 20000
#define NEIGH  12
#define HDIM   64
#define NCOL   256      // [root | rel0 | rel1 | rel2] x 64
#define TILE_M 64
#define NEDGE  (NNODES * NEIGH)

// ---------------------------------------------------------------------------
// Pack f32 weights (Wroot: K x 64, Wrel: 3 x K x 64) into f16 in this file's (contiguous-k)
// WMMA B-fragment layout: frag = (kstep*16 + colfrag), within frag:
// lane l, half h  <->  k = 32*kstep + h + (l<16 ? 0 : 16), col = 16*colfrag + l%16
// ---------------------------------------------------------------------------
__global__ __launch_bounds__(256) void pack_weights(
    const float* __restrict__ Wroot, const float* __restrict__ Wrel,
    _Float16* __restrict__ Bsw, int K)
{
  int idx = blockIdx.x * 256 + threadIdx.x;
  if (idx >= K * NCOL) return;
  int t   = idx >> 13;
  int rem = idx & 8191;
  int n   = rem >> 9;
  int w   = rem & 511;
  int l   = w >> 4;
  int h   = w & 15;
  int k   = 32 * t + h + ((l < 16) ? 0 : 16);
  int col = n * 16 + (l & 15);
  float v;
  if (col < HDIM) {
    v = Wroot[k * HDIM + col];
  } else {
    int r = (col - HDIM) >> 6;
    v = Wrel[(r * K + k) * HDIM + (col & 63)];
  }
  Bsw[idx] = (_Float16)v;
}

// ---------------------------------------------------------------------------
// Fused RGCN projection GEMM:  Y(N x 256) = A(N x K) @ B(K x 256)
// MODE 0: A[i][k] = exp(-(raw[i][k/S] - filt[k%S])^2 / gamma^2)  (gbf expand)
// MODE 1: A[i][k] = relu(raw[i][k])                              (hidden, G==K)
// Block: 256 thr = 8 waves (2 x 4). Tile 64 rows x 256 cols.
// Pipelined: double-buffered A staging in LDS, register-prefetched B.
// ---------------------------------------------------------------------------
template <int MODE, int G, int S>
__global__ __launch_bounds__(256) void gemm_rgcn(
    const float* __restrict__ raw, float dmin, float dmax,
    const _Float16* __restrict__ Bsw, float* __restrict__ Y)
{
  constexpr int K   = (MODE == 0) ? G * S : G;
  constexpr int nKT = K >> 5;
  __shared__ __align__(16) float    rawT[TILE_M * G];
  __shared__ __align__(32) _Float16 aT[2][4 * 512];

  const int tid  = threadIdx.x;
  const int lane = tid & 31;
  const int wave = tid >> 5;
  const int wM   = wave & 1;    // row half (32 rows each)
  const int wN   = wave >> 1;   // 64-col group (0..3)
  const int row0 = blockIdx.x * TILE_M;

  // stage raw rows (zero-pad past N)
  for (int i = tid; i < TILE_M * G; i += 256) {
    int r = row0 + i / G;
    rawT[i] = (r < NNODES) ? raw[(size_t)row0 * G + i] : 0.f;
  }

  const float scale = (S > 1) ? (dmax - dmin) / (float)(S - 1) : 0.f;
  const float gamma = (dmax - dmin) / (float)S;
  const float ig2   = (MODE == 0) ? 1.f / (gamma * gamma) : 0.f;
  const float cneg  = -ig2 * 1.44269504088896f;   // fold 1/g^2 and log2(e)
  const float Sf    = (float)S;

  // generator: this thread produces 8 halves with CONSECUTIVE k
  // (one ds_store_b128; the 8 k span at most two filter groups g)
  const int base  = tid * 8;
  const int fr    = base >> 9;          // fragment 0..3
  const int gl    = (base >> 4) & 31;   // lane within fragment
  const int h0    = base & 15;          // 0 or 8
  const int rowl  = fr * 16 + (gl & 15);
  const int kofs  = (gl >= 16) ? 16 : 0;
  const int kbase = kofs + h0;  // kg = kt*32 + kbase + u

  auto genA = [&](int kt, int buf) {
    _Float16 tmp[8];
    const int kg0 = (kt << 5) + kbase;
    if (MODE == 0) {
      int g0 = kg0 / S;                 // single const-divisor division
      int s0 = kg0 - g0 * S;
      int g1 = (g0 + 1 < G) ? g0 + 1 : g0;
      float x0  = rawT[rowl * G + g0];
      float x1  = rawT[rowl * G + g1];
      float s0f = (float)s0;
      float e0  = x0 - (dmin + scale * s0f);        // d(u) = e0 - scale*u
      float e1  = x1 - (dmin + scale * (s0f - Sf)); // after wrap
      int   thr = S - s0;                            // wrap when u >= thr
      #pragma unroll
      for (int u = 0; u < 8; ++u) {
        float d   = ((u >= thr) ? e1 : e0) - scale * (float)u;
        float arg = (d * d) * cneg;
        tmp[u] = (_Float16)__builtin_amdgcn_exp2f(arg);  // raw v_exp_f32
      }
    } else {
      const float* rp = &rawT[rowl * G + kg0];   // 32B-aligned, contiguous
      #pragma unroll
      for (int u = 0; u < 8; ++u) {
        float v = rp[u];
        tmp[u] = (_Float16)(v > 0.f ? v : 0.f);  // relu applied at consumption
      }
    }
    *(v8h*)&aT[buf][base] = *(const v8h*)tmp;
  };

  __syncthreads();       // rawT ready
  genA(0, 0);

  v8f acc0 = {}, acc1 = {}, acc2 = {}, acc3 = {};
  v8f acc4 = {}, acc5 = {}, acc6 = {}, acc7 = {};

  // B fragment pointers: frag stride = 32 v16h, kstep stride = 512 v16h
  const v16h* bb = (const v16h*)Bsw + (size_t)(wN * 4) * 32 + lane;
  v16h nb0 = bb[0], nb1 = bb[32], nb2 = bb[64], nb3 = bb[96];

  __syncthreads();       // aT[0] ready
  int buf = 0;

  #pragma unroll 2
  for (int kt = 0; kt < nKT; ++kt) {
    v16h b0 = nb0, b1 = nb1, b2 = nb2, b3 = nb3;
    if (kt + 1 < nKT) {                       // prefetch next B (hidden latency)
      const v16h* p = bb + (size_t)(kt + 1) * 512;
      nb0 = p[0]; nb1 = p[32]; nb2 = p[64]; nb3 = p[96];
    }
    // current A fragments (issued before the generator's ds_stores)
    v16h a0 = *((const v16h*)(aT[buf] + (wM * 2 + 0) * 512) + lane);
    v16h a1 = *((const v16h*)(aT[buf] + (wM * 2 + 1) * 512) + lane);
    // generate next iteration's A into the other buffer
    if (kt + 1 < nKT) genA(kt + 1, buf ^ 1);

    acc0 = __builtin_amdgcn_wmma_f32_16x16x32_f16(false, a0, false, b0, (short)0, acc0, false, false);
    acc1 = __builtin_amdgcn_wmma_f32_16x16x32_f16(false, a0, false, b1, (short)0, acc1, false, false);
    acc2 = __builtin_amdgcn_wmma_f32_16x16x32_f16(false, a0, false, b2, (short)0, acc2, false, false);
    acc3 = __builtin_amdgcn_wmma_f32_16x16x32_f16(false, a0, false, b3, (short)0, acc3, false, false);
    acc4 = __builtin_amdgcn_wmma_f32_16x16x32_f16(false, a1, false, b0, (short)0, acc4, false, false);
    acc5 = __builtin_amdgcn_wmma_f32_16x16x32_f16(false, a1, false, b1, (short)0, acc5, false, false);
    acc6 = __builtin_amdgcn_wmma_f32_16x16x32_f16(false, a1, false, b2, (short)0, acc6, false, false);
    acc7 = __builtin_amdgcn_wmma_f32_16x16x32_f16(false, a1, false, b3, (short)0, acc7, false, false);

    __syncthreads();     // next A buffer complete; this buffer free to rewrite
    buf ^= 1;
  }

  // ---- write C: VGPR v -> row = v + (lane<16?0:8); col = lane%16 ----
  const int rb0   = row0 + wM * 32 + ((lane < 16) ? 0 : 8);
  const int cbase = wN * 64 + (lane & 15);
  #pragma unroll
  for (int v = 0; v < 8; ++v) {
    int ra = rb0 + v;
    if (ra < NNODES) {
      float* yr = Y + (size_t)ra * NCOL + cbase;
      yr[0]  = acc0[v]; yr[16] = acc1[v]; yr[32] = acc2[v]; yr[48] = acc3[v];
    }
    int rc = rb0 + 16 + v;
    if (rc < NNODES) {
      float* yr = Y + (size_t)rc * NCOL + cbase;
      yr[0]  = acc4[v]; yr[16] = acc5[v]; yr[32] = acc6[v]; yr[48] = acc7[v];
    }
  }
}

// ---------------------------------------------------------------------------
// Reverse-CSR build (once) + atomic-free gather aggregation (per layer)
// ---------------------------------------------------------------------------
__global__ void zero_i32(int* __restrict__ p, int n) {
  int i = blockIdx.x * 256 + threadIdx.x;
  if (i < n) p[i] = 0;
}

__global__ void count_edges(const int* __restrict__ species,
                            const int* __restrict__ nbr, int* __restrict__ cnt) {
  int e = blockIdx.x * 256 + threadIdx.x;
  if (e >= NEDGE) return;
  int i = e / NEIGH;
  int d = nbr[e];
  int r = species[i] + species[d];
  atomicAdd(cnt + r * NNODES + d, 1);
}

__global__ void inv_counts(const int* __restrict__ cnt, float* __restrict__ invc) {
  int i = blockIdx.x * 256 + threadIdx.x;
  if (i >= 3 * NNODES) return;
  int c = cnt[i];
  invc[i] = 1.f / (float)(c > 1 ? c : 1);
}

// single-block exclusive scan of total in-degree -> off[N+1], pos[N]=off[N]
__global__ __launch_bounds__(256) void scan_offsets(
    const int* __restrict__ cnt, int* __restrict__ off, int* __restrict__ pos)
{
  __shared__ int part[256];
  const int CH = (NNODES + 255) / 256;   // 79
  int t  = threadIdx.x;
  int s0 = t * CH;
  int e0 = (s0 + CH < NNODES) ? s0 + CH : NNODES;
  int sum = 0;
  for (int i = s0; i < e0; ++i)
    sum += cnt[i] + cnt[NNODES + i] + cnt[2 * NNODES + i];
  part[t] = sum;
  __syncthreads();
  for (int d = 1; d < 256; d <<= 1) {    // inclusive scan of partials
    int v = (t >= d) ? part[t - d] : 0;
    __syncthreads();
    part[t] += v;
    __syncthreads();
  }
  int run = (t == 0) ? 0 : part[t - 1];
  for (int i = s0; i < e0; ++i) {
    off[i] = run;
    pos[i] = run;
    run += cnt[i] + cnt[NNODES + i] + cnt[2 * NNODES + i];
  }
  if (t == 255) off[NNODES] = run;
}

// fill packed incoming-edge records: edges[slot] = (src<<2) | rel
__global__ void fill_edges(const int* __restrict__ species,
                           const int* __restrict__ nbr,
                           int* __restrict__ pos, int* __restrict__ edges) {
  int e = blockIdx.x * 256 + threadIdx.x;
  if (e >= NEDGE) return;
  int i = e / NEIGH;
  int d = nbr[e];
  int r = species[i] + species[d];
  int slot = atomicAdd(pos + d, 1);
  edges[slot] = (i << 2) | r;
}

// out[i][h] = Y_root[i][h] + bias[h] + sum over incoming edges (pure gather)
__global__ void aggregate(const float* __restrict__ Y, const float* __restrict__ bias,
                          const int* __restrict__ off, const int* __restrict__ edges,
                          const float* __restrict__ invc, float* __restrict__ out) {
  int idx = blockIdx.x * 256 + threadIdx.x;
  if (idx >= NNODES * HDIM) return;
  int i = idx >> 6, h = idx & 63;
  float acc = Y[(size_t)i * NCOL + h] + bias[h];
  int b = off[i], e = off[i + 1];
  for (int t = b; t < e; ++t) {
    int p = edges[t];
    int r = p & 3, s = p >> 2;
    acc += Y[(size_t)s * NCOL + HDIM + r * HDIM + h] * invc[r * NNODES + i];
  }
  out[idx] = acc;
}

// per-crystal mean pool of relu([bf2|af2]) then 128->2 FC
__global__ __launch_bounds__(256) void pool_fc(
    const float* __restrict__ bf, const float* __restrict__ af,
    const int* __restrict__ crys, const float* __restrict__ fcW,
    const float* __restrict__ fcb, float* __restrict__ outp)
{
  __shared__ float sums[2][128];
  int cr = blockIdx.x;
  int s = crys[cr * 2], e = crys[cr * 2 + 1];
  int f    = threadIdx.x & 127;
  int half = threadIdx.x >> 7;
  const float* src = (f < 64) ? bf : af;
  int h = f & 63;
  float acc = 0.f;
  for (int i = s + half; i < e; i += 2) {
    float v = src[(size_t)i * 64 + h];
    acc += v > 0.f ? v : 0.f;
  }
  sums[half][f] = acc;
  __syncthreads();
  if (threadIdx.x < 2) {
    float o = fcb[threadIdx.x];
    float invn = 1.f / (float)(e - s);
    for (int ff = 0; ff < 128; ++ff)
      o += (sums[0][ff] + sums[1][ff]) * invn * fcW[ff * 2 + threadIdx.x];
    outp[cr * 2 + threadIdx.x] = o;
  }
}

// ---------------------------------------------------------------------------
extern "C" void kernel_launch(void* const* d_in, const int* in_sizes, int n_in,
                              void* d_out, int out_size, void* d_ws, size_t ws_size,
                              hipStream_t stream) {
  (void)in_sizes; (void)n_in; (void)out_size; (void)ws_size;
  const float* bond     = (const float*)d_in[0];
  const float* angle    = (const float*)d_in[1];
  const int*   species  = (const int*)d_in[2];
  const int*   nbr      = (const int*)d_in[3];
  const int*   crys     = (const int*)d_in[4];
  const float* W1b_rel  = (const float*)d_in[5];
  const float* W1b_root = (const float*)d_in[6];
  const float* b1b      = (const float*)d_in[7];
  const float* W1a_rel  = (const float*)d_in[8];
  const float* W1a_root = (const float*)d_in[9];
  const float* b1a      = (const float*)d_in[10];
  const float* W2b_rel  = (const float*)d_in[11];
  const float* W2b_root = (const float*)d_in[12];
  const float* b2b      = (const float*)d_in[13];
  const float* W2a_rel  = (const float*)d_in[14];
  const float* W2a_root = (const float*)d_in[15];
  const float* b2a      = (const float*)d_in[16];
  const float* fcW      = (const float*)d_in[17];
  const float* fcb      = (const float*)d_in[18];
  float* out = (float*)d_out;

  char* wp = (char*)d_ws;
  auto alloc = [&](size_t bytes) -> char* {
    char* p = wp;
    wp += (bytes + 255) & ~(size_t)255;
    return p;
  };
  _Float16* Bb1 = (_Float16*)alloc((size_t)480 * NCOL * 2);
  _Float16* Ba1 = (_Float16*)alloc((size_t)2880 * NCOL * 2);
  _Float16* Bb2 = (_Float16*)alloc((size_t)64 * NCOL * 2);
  _Float16* Ba2 = (_Float16*)alloc((size_t)64 * NCOL * 2);
  float* Y    = (float*)alloc((size_t)NNODES * NCOL * 4);
  float* bf1  = (float*)alloc((size_t)NNODES * HDIM * 4);
  float* af1  = (float*)alloc((size_t)NNODES * HDIM * 4);
  float* bf2  = (float*)alloc((size_t)NNODES * HDIM * 4);
  float* af2  = (float*)alloc((size_t)NNODES * HDIM * 4);
  int*   cnt   = (int*)alloc((size_t)3 * NNODES * 4);
  float* invc  = (float*)alloc((size_t)3 * NNODES * 4);
  int*   off   = (int*)alloc((size_t)(NNODES + 1) * 4);
  int*   pos   = (int*)alloc((size_t)NNODES * 4);
  int*   edges = (int*)alloc((size_t)NEDGE * 4);

  // ---- static prep: packed f16 weights + reverse CSR + inverse counts ----
  pack_weights<<<480,  256, 0, stream>>>(W1b_root, W1b_rel, Bb1, 480);
  pack_weights<<<2880, 256, 0, stream>>>(W1a_root, W1a_rel, Ba1, 2880);
  pack_weights<<<64,   256, 0, stream>>>(W2b_root, W2b_rel, Bb2, 64);
  pack_weights<<<64,   256, 0, stream>>>(W2a_root, W2a_rel, Ba2, 64);

  zero_i32<<<(3 * NNODES + 255) / 256, 256, 0, stream>>>(cnt, 3 * NNODES);
  count_edges<<<(NEDGE + 255) / 256, 256, 0, stream>>>(species, nbr, cnt);
  inv_counts<<<(3 * NNODES + 255) / 256, 256, 0, stream>>>(cnt, invc);
  scan_offsets<<<1, 256, 0, stream>>>(cnt, off, pos);
  fill_edges<<<(NEDGE + 255) / 256, 256, 0, stream>>>(species, nbr, pos, edges);

  const int TILES = (NNODES + TILE_M - 1) / TILE_M;   // 313
  const int NB64  = (NNODES * HDIM) / 256;            // 5000 (exact)

  // ---- layer 1, bond: gbf(0..8, 40 steps), K=480 ----
  gemm_rgcn<0, 12, 40><<<TILES, 256, 0, stream>>>(bond, 0.f, 8.f, Bb1, Y);
  aggregate<<<NB64, 256, 0, stream>>>(Y, b1b, off, edges, invc, bf1);

  // ---- layer 1, angle: gbf(-1..1, 20 steps), K=2880 ----
  gemm_rgcn<0, 144, 20><<<TILES, 256, 0, stream>>>(angle, -1.f, 1.f, Ba1, Y);
  aggregate<<<NB64, 256, 0, stream>>>(Y, b1a, off, edges, invc, af1);

  // ---- layer 2, bond: input relu(bf1), K=64 ----
  gemm_rgcn<1, 64, 1><<<TILES, 256, 0, stream>>>(bf1, 0.f, 0.f, Bb2, Y);
  aggregate<<<NB64, 256, 0, stream>>>(Y, b2b, off, edges, invc, bf2);

  // ---- layer 2, angle: input relu(af1), K=64 ----
  gemm_rgcn<1, 64, 1><<<TILES, 256, 0, stream>>>(af1, 0.f, 0.f, Ba2, Y);
  aggregate<<<NB64, 256, 0, stream>>>(Y, b2a, off, edges, invc, af2);

  // ---- mean pool per crystal + FC ----
  pool_fc<<<20, 256, 0, stream>>>(bf2, af2, crys, fcW, fcb, out);
}
